// MultiHeadAttention_47072841564482
// MI455X (gfx1250) — matrix operand order repaired and checked by reading, not compiled, not run on hardware
//
#include <hip/hip_runtime.h>

typedef _Float16 half_t;
typedef __attribute__((ext_vector_type(16))) _Float16     v16h;
typedef __attribute__((ext_vector_type(8)))  float        v8f;
typedef __attribute__((ext_vector_type(4)))  unsigned int v4u;
typedef __attribute__((ext_vector_type(2)))  unsigned int v2u;

union FragA { v16h v; unsigned int u[8]; };
union FragB { v16h v; v4u q[2]; v2u d[4]; };

// ---- CDNA5 async global->LDS copy (ASYNCcnt path), with guarded fallback ---
#if defined(__has_builtin)
#if __has_builtin(__builtin_amdgcn_global_load_async_to_lds_b128)
#define HAVE_ASYNC_LDS 1
#endif
#endif

#ifdef HAVE_ASYNC_LDS
// Builtin signature (from clang diagnostic): param0 is
// 'int __vector(4) __device__ *' (AS1), param1 the LDS-side pointer (AS3).
typedef int v4i_t __attribute__((__vector_size__(16)));
typedef __attribute__((address_space(1))) v4i_t* gptr128_t;
typedef __attribute__((address_space(3))) v4i_t* lptr128_t;
#define ASYNC_CP128(dst_lds, src_glob)                                        \
  __builtin_amdgcn_global_load_async_to_lds_b128((gptr128_t)(src_glob),       \
                                                 (lptr128_t)(dst_lds), 0, 0)
#if __has_builtin(__builtin_amdgcn_s_wait_asynccnt)
#define WAIT_ASYNC(n) __builtin_amdgcn_s_wait_asynccnt(n)
#else
#define WAIT_ASYNC(n) asm volatile("s_wait_asynccnt %0" ::"i"(n) : "memory")
#endif
#endif

__device__ inline v8f wmma16(v16h a, v16h b, v8f c) {
  return __builtin_amdgcn_wmma_f32_16x16x32_f16(false, a, false, b, (short)0, c,
                                                false, false);
}

__device__ inline v8f vzero() {
  v8f z = {0.f, 0.f, 0.f, 0.f, 0.f, 0.f, 0.f, 0.f};
  return z;
}

// A-matrix (16x32 f16) per-lane K index for half-element h (0..15):
// lanes 0-15 hold k in {0..15}; lanes 16-31 hold {16..31}.
__device__ inline int a_k(int lane, int h) {
  return (lane < 16) ? h : h + 16;
}

// ---------------------------------------------------------------------------
// Precision conversion / weight transpose kernels
// ---------------------------------------------------------------------------
__global__ __launch_bounds__(256) void f32_to_f16_kernel(
    const float* __restrict__ src, half_t* __restrict__ dst, int n) {
  int idx = blockIdx.x * 256 + threadIdx.x;
  if (idx < n) dst[idx] = (half_t)src[idx];
}

// src is [K][N] row-major f32; dst is [N][K] row-major f16
__global__ __launch_bounds__(256) void transpose_f32_to_f16_kernel(
    const float* __restrict__ src, half_t* __restrict__ dst, int K, int N) {
  int idx = blockIdx.x * 256 + threadIdx.x;
  if (idx >= K * N) return;
  int n = idx / K;
  int k = idx - n * K;
  dst[idx] = (half_t)src[(size_t)k * N + n];
}

// ---------------------------------------------------------------------------
// Shared WMMA GEMM mainloop: C[128x128] block tile, 256 threads (8 waves in
// a 4(M) x 2(N) wave grid; each wave owns 32(M) x 64(N) = 2x4 WMMA tiles).
// A: [M][Ksize] f16 row-major. Bt: [N][Ksize] f16 row-major (B transposed).
// Double-buffered LDS; async global->LDS copies when available.
// ---------------------------------------------------------------------------
#define LDK 48  // 32 k-halfs + 16 pad -> 96B row stride (16B aligned)

__device__ inline void gemm_compute_tile(const half_t (*Al)[LDK],
                                         const half_t (*Bl)[LDK], int lane,
                                         int wr, int wc, v8f acc[8]) {
  FragA af[2];
#pragma unroll
  for (int s = 0; s < 2; ++s) {
    int r = wr * 32 + s * 16 + (lane & 15);
#pragma unroll
    for (int p = 0; p < 8; ++p)
      af[s].u[p] = *(const unsigned int*)&Al[r][a_k(lane, 2 * p)];
  }
  const int koff = (lane < 16) ? 0 : 16;
  FragB bf[4];
#pragma unroll
  for (int t = 0; t < 4; ++t) {
    int n = wc * 64 + t * 16 + (lane & 15);
    bf[t].q[0] = *(const v4u*)&Bl[n][koff];
    bf[t].q[1] = *(const v4u*)&Bl[n][koff + 8];
  }
#pragma unroll
  for (int s = 0; s < 2; ++s)
#pragma unroll
    for (int t = 0; t < 4; ++t)
      acc[s * 4 + t] = wmma16(af[s].v, bf[t].v, acc[s * 4 + t]);
}

__device__ inline void gemm_tiles(const half_t* __restrict__ A,
                                  const half_t* __restrict__ Bt, int Ksize,
                                  int mBlock, int nBlock,
                                  half_t (*Al)[128][LDK],  // [2][128][LDK]
                                  half_t (*Bl)[128][LDK],  // [2][128][LDK]
                                  v8f acc[8]) {
  const int tid = threadIdx.x, lane = tid & 31, wave = tid >> 5;
  const int wr = wave & 3, wc = wave >> 2;
  const int row = tid >> 1, seg = (tid & 1) * 16;
#pragma unroll
  for (int i = 0; i < 8; ++i) acc[i] = vzero();

  const half_t* arow = A + (size_t)(mBlock * 128 + row) * Ksize + seg;
  const half_t* brow = Bt + (size_t)(nBlock * 128 + row) * Ksize + seg;

#ifdef HAVE_ASYNC_LDS
  // ---- async double-buffered pipeline: 1 barrier + 1 wait per K-step ----
  auto issue = [&](int k0, int b) {
    ASYNC_CP128(&Al[b][row][seg], arow + k0);
    ASYNC_CP128(&Al[b][row][seg + 8], arow + k0 + 8);
    ASYNC_CP128(&Bl[b][row][seg], brow + k0);
    ASYNC_CP128(&Bl[b][row][seg + 8], brow + k0 + 8);
  };
  issue(0, 0);
  WAIT_ASYNC(0);
  __syncthreads();
  int buf = 0;
  for (int k0 = 0; k0 < Ksize; k0 += 32) {
    if (k0 + 32 < Ksize) issue(k0 + 32, buf ^ 1);  // prefetch next tile
    gemm_compute_tile(Al[buf], Bl[buf], lane, wr, wc, acc);
    WAIT_ASYNC(0);
    __syncthreads();
    buf ^= 1;
  }
#else
  // ---- fallback: VGPR-staged, single buffer, two barriers per K-step ----
  for (int k0 = 0; k0 < Ksize; k0 += 32) {
    v4u a0 = *(const v4u*)(arow + k0);
    v4u a1 = *(const v4u*)(arow + k0 + 8);
    v4u b0 = *(const v4u*)(brow + k0);
    v4u b1 = *(const v4u*)(brow + k0 + 8);
    __syncthreads();
    *(v4u*)&Al[0][row][seg] = a0;
    *(v4u*)&Al[0][row][seg + 8] = a1;
    *(v4u*)&Bl[0][row][seg] = b0;
    *(v4u*)&Bl[0][row][seg + 8] = b1;
    __syncthreads();
    gemm_compute_tile(Al[0], Bl[0], lane, wr, wc, acc);
  }
#endif
}

// ---------------------------------------------------------------------------
// GEMM1: qkv = x @ Wqkv + bqkv, scattered to Q/K/V [b*h][2048][64] f16,
// with 1/sqrt(d) folded into Q.
// ---------------------------------------------------------------------------
__global__ __launch_bounds__(256) void gemm_qkv_kernel(
    const half_t* __restrict__ xh, const half_t* __restrict__ WqkvT,
    const float* __restrict__ bqkv, half_t* __restrict__ Qb,
    half_t* __restrict__ Kb, half_t* __restrict__ Vb) {
  __shared__ half_t Al[2][128][LDK];
  __shared__ half_t Bl[2][128][LDK];
  v8f acc[8];
  gemm_tiles(xh, WqkvT, 1024, blockIdx.x, blockIdx.y, Al, Bl, acc);

  const int lane = threadIdx.x & 31, wave = threadIdx.x >> 5;
  const int wr = wave & 3, wc = wave >> 2;
  const int hi = (lane >> 4) * 8;  // C layout: vgpr j -> row j (lo) / j+8 (hi)
  const int nl = lane & 15;
#pragma unroll
  for (int s = 0; s < 2; ++s) {
#pragma unroll
    for (int t = 0; t < 4; ++t) {
      int n = blockIdx.y * 128 + wc * 64 + t * 16 + nl;
      int chunk = n >> 10;  // 0=q 1=k 2=v
      int c = n & 1023;
      int head = c >> 6, dcol = c & 63;
      half_t* dst = (chunk == 0) ? Qb : (chunk == 1) ? Kb : Vb;
      float scale = (chunk == 0) ? 0.125f : 1.0f;  // 1/sqrt(64)
      float bias = bqkv[n];
#pragma unroll
      for (int j = 0; j < 8; ++j) {
        int m = blockIdx.x * 128 + wr * 32 + s * 16 + j + hi;
        int b = m >> 11, r = m & 2047;
        float v = (acc[s * 4 + t][j] + bias) * scale;
        dst[((size_t)((b * 16 + head) * 2048 + r)) * 64 + dcol] = (half_t)v;
      }
    }
  }
}

// ---------------------------------------------------------------------------
// Flash attention: grid (b*h, 2048/128). Each wave owns a 16-row Q tile,
// streams 32-key K/V tiles through LDS, online softmax, O += P@V via WMMA.
// K tile goes through the async global->LDS path when available.
// ---------------------------------------------------------------------------
__global__ __launch_bounds__(256) void attn_kernel(const half_t* __restrict__ Qb,
                                                   const half_t* __restrict__ Kb,
                                                   const half_t* __restrict__ Vb,
                                                   half_t* __restrict__ Oh) {
  __shared__ half_t Kt[32][72];     // [key][d]  (row = Bt row for S)
  __shared__ half_t Vt[64][40];     // [d][key]  (transposed -> Bt rows for PV)
  __shared__ half_t Pl[8][16][40];  // per-wave P staging (C-layout -> A-layout)

  const int bh = blockIdx.x, qc = blockIdx.y;
  const int tid = threadIdx.x, lane = tid & 31, wave = tid >> 5;
  const int nl = lane & 15;
  const int koff = (lane < 16) ? 0 : 16;
  const int qbase = qc * 128 + wave * 16;
  const size_t bh_off = (size_t)bh * 2048 * 64;

  // Q A-fragments (scale already folded in): 16x64 = two 16x32 fragments
  FragA qa[2];
  {
    const half_t* qrow = Qb + bh_off + (size_t)(qbase + nl) * 64;
#pragma unroll
    for (int f = 0; f < 2; ++f)
#pragma unroll
      for (int p = 0; p < 8; ++p)
        qa[f].u[p] = *(const unsigned int*)(qrow + f * 32 + a_k(lane, 2 * p));
  }

  v8f o[4];
#pragma unroll
  for (int t = 0; t < 4; ++t) o[t] = vzero();
  float mrow[8], lsum[8];
#pragma unroll
  for (int j = 0; j < 8; ++j) { mrow[j] = -3.0e38f; lsum[j] = 0.f; }

  const int ldr = tid >> 3;       // 0..31 (key row)
  const int ldc = (tid & 7) * 8;  // 0..56 (d col)

  for (int kb = 0; kb < 2048; kb += 32) {
    const half_t* krow = Kb + bh_off + (size_t)(kb + ldr) * 64 + ldc;
    v4u vv = *(const v4u*)(Vb + bh_off + (size_t)(kb + ldr) * 64 + ldc);
#ifndef HAVE_ASYNC_LDS
    v4u kv = *(const v4u*)krow;
#endif
    __syncthreads();  // previous tile fully consumed
#ifdef HAVE_ASYNC_LDS
    ASYNC_CP128(&Kt[ldr][ldc], krow);
#else
    *(v4u*)&Kt[ldr][ldc] = kv;
#endif
    {
      union { v4u q; half_t h[8]; } tmp;
      tmp.q = vv;
#pragma unroll
      for (int i = 0; i < 8; ++i) Vt[ldc + i][ldr] = tmp.h[i];
    }
#ifdef HAVE_ASYNC_LDS
    WAIT_ASYNC(0);
#endif
    __syncthreads();

    // S = Q @ K^T : two 16x16 tiles (keys 0-15 -> s0, keys 16-31 -> s1)
    v8f s0 = vzero(), s1 = vzero();
    FragB kf;
#pragma unroll
    for (int f = 0; f < 2; ++f) {
      kf.q[0] = *(const v4u*)&Kt[nl][f * 32 + koff];
      kf.q[1] = *(const v4u*)&Kt[nl][f * 32 + koff + 8];
      s0 = wmma16(qa[f].v, kf.v, s0);
    }
#pragma unroll
    for (int f = 0; f < 2; ++f) {
      kf.q[0] = *(const v4u*)&Kt[16 + nl][f * 32 + koff];
      kf.q[1] = *(const v4u*)&Kt[16 + nl][f * 32 + koff + 8];
      s1 = wmma16(qa[f].v, kf.v, s1);
    }

    // Online softmax. Row M=j(+8) lives on 16 lanes of a half-wave; xor masks
    // 1,2,4,8 stay within the half.
#pragma unroll
    for (int j = 0; j < 8; ++j) {
      float a = fmaxf(s0[j], s1[j]);
#pragma unroll
      for (int msk = 1; msk < 16; msk <<= 1) a = fmaxf(a, __shfl_xor(a, msk, 32));
      float mn = fmaxf(mrow[j], a);
      float corr = __expf(mrow[j] - mn);
      float p0 = __expf(s0[j] - mn);
      float p1 = __expf(s1[j] - mn);
      float rs = p0 + p1;
#pragma unroll
      for (int msk = 1; msk < 16; msk <<= 1) rs += __shfl_xor(rs, msk, 32);
      lsum[j] = lsum[j] * corr + rs;
      mrow[j] = mn;
#pragma unroll
      for (int t = 0; t < 4; ++t) o[t][j] *= corr;
      s0[j] = p0;
      s1[j] = p1;
    }

    // Stage P (16x32 f16) in wave-private LDS (C-layout write)
    const int prow = (lane >> 4) * 8;
#pragma unroll
    for (int j = 0; j < 8; ++j) {
      Pl[wave][prow + j][nl] = (half_t)s0[j];
      Pl[wave][prow + j][nl + 16] = (half_t)s1[j];
    }
    // Re-read as A-fragment (same-wave DS ops are in-order)
    FragA pf;
    {
      const half_t* pr = &Pl[wave][nl][0];
#pragma unroll
      for (int p = 0; p < 8; ++p)
        pf.u[p] = *(const unsigned int*)(pr + a_k(lane, 2 * p));
    }

    // O += P @ V : four 16-wide d-column tiles
#pragma unroll
    for (int t = 0; t < 4; ++t) {
      FragB vf;
      const half_t* vr = &Vt[t * 16 + nl][koff];
      vf.d[0] = *(const v2u*)(vr);
      vf.d[1] = *(const v2u*)(vr + 4);
      vf.d[2] = *(const v2u*)(vr + 8);
      vf.d[3] = *(const v2u*)(vr + 12);
      o[t] = wmma16(pf.v, vf.v, o[t]);
    }
  }

  // Normalize and write O back in [b][n][h*64+d] f16 layout for GEMM2
  const int prow = (lane >> 4) * 8;
  const int b = bh >> 4, head = bh & 15;
#pragma unroll
  for (int j = 0; j < 8; ++j) {
    float inv = 1.0f / lsum[j];
    int nrow = qbase + prow + j;
    size_t base = ((size_t)(b * 2048 + nrow)) * 1024 + head * 64;
#pragma unroll
    for (int t = 0; t < 4; ++t) Oh[base + t * 16 + nl] = (half_t)(o[t][j] * inv);
  }
}

// ---------------------------------------------------------------------------
// GEMM2: out = O @ Wo + bo (f32 output)
// ---------------------------------------------------------------------------
__global__ __launch_bounds__(256) void gemm_out_kernel(
    const half_t* __restrict__ Oh, const half_t* __restrict__ WoT,
    const float* __restrict__ bo, float* __restrict__ out) {
  __shared__ half_t Al[2][128][LDK];
  __shared__ half_t Bl[2][128][LDK];
  v8f acc[8];
  gemm_tiles(Oh, WoT, 1024, blockIdx.x, blockIdx.y, Al, Bl, acc);

  const int lane = threadIdx.x & 31, wave = threadIdx.x >> 5;
  const int wr = wave & 3, wc = wave >> 2;
  const int hi = (lane >> 4) * 8;
  const int nl = lane & 15;
#pragma unroll
  for (int s = 0; s < 2; ++s) {
#pragma unroll
    for (int t = 0; t < 4; ++t) {
      int n = blockIdx.y * 128 + wc * 64 + t * 16 + nl;
      float bias = bo[n];
#pragma unroll
      for (int j = 0; j < 8; ++j) {
        int m = blockIdx.x * 128 + wr * 32 + s * 16 + j + hi;
        out[(size_t)m * 1024 + n] = acc[s * 4 + t][j] + bias;
      }
    }
  }
}

// ---------------------------------------------------------------------------
extern "C" void kernel_launch(void* const* d_in, const int* in_sizes, int n_in,
                              void* d_out, int out_size, void* d_ws,
                              size_t ws_size, hipStream_t stream) {
  (void)in_sizes; (void)n_in; (void)out_size; (void)ws_size;
  const float* x    = (const float*)d_in[0];
  const float* Wqkv = (const float*)d_in[1];
  const float* bqkv = (const float*)d_in[2];
  const float* Wo   = (const float*)d_in[3];
  const float* bo   = (const float*)d_in[4];
  float* out = (float*)d_out;

  char* ws = (char*)d_ws;
  half_t* xh    = (half_t*)ws; ws += (size_t)8192 * 1024 * 2;  // x as f16
  half_t* WqkvT = (half_t*)ws; ws += (size_t)3072 * 1024 * 2;  // Wqkv^T f16
  half_t* WoT   = (half_t*)ws; ws += (size_t)1024 * 1024 * 2;  // Wo^T f16
  half_t* Qb    = (half_t*)ws; ws += (size_t)64 * 2048 * 64 * 2;
  half_t* Kb    = (half_t*)ws; ws += (size_t)64 * 2048 * 64 * 2;
  half_t* Vb    = (half_t*)ws; ws += (size_t)64 * 2048 * 64 * 2;
  half_t* Oh    = (half_t*)ws;  // attn output f16

  f32_to_f16_kernel<<<32768, 256, 0, stream>>>(x, xh, 8388608);
  transpose_f32_to_f16_kernel<<<12288, 256, 0, stream>>>(Wqkv, WqkvT, 1024, 3072);
  transpose_f32_to_f16_kernel<<<4096, 256, 0, stream>>>(Wo, WoT, 1024, 1024);
  gemm_qkv_kernel<<<dim3(64, 24), 256, 0, stream>>>(xh, WqkvT, bqkv, Qb, Kb, Vb);
  attn_kernel<<<dim3(64, 16), 256, 0, stream>>>(Qb, Kb, Vb, Oh);
  gemm_out_kernel<<<dim3(64, 8), 256, 0, stream>>>(Oh, WoT, bo, out);
}
